// RecurrentGCN_82514911690903
// MI455X (gfx1250) — matrix operand order repaired and checked by reading, not compiled, not run on hardware
//
#include <hip/hip_runtime.h>
#include <hip/hip_bf16.h>
#include <math.h>

// ---------------------------------------------------------------------------
// RecurrentGCN (MPNNLSTM window=1, eval) for MI455X / gfx1250, wave32.
//   N=100000 nodes, E=3.2M edges, F_IN=8, H=64.
// Dense GEMM stages: V_WMMA_F32_16X16X32_F16 (f16 in, f32 accum), K templated;
// fragment-edge guards are data-selects (cndmask), not EXEC diamonds.
// Edge propagation: one wave per edge, f32 atomics into L2-resident agg.
// ---------------------------------------------------------------------------

typedef __attribute__((ext_vector_type(16))) _Float16 v16h;
typedef __attribute__((ext_vector_type(8)))  float    v8f;

#define NNODES 100000
#define NH     64
#define FINN   8
#define BN_EPS 1e-5f

// ---------------------------------------------------------------- utilities
__global__ void set_value_kernel(float* __restrict__ p, float v, int n) {
  int i = blockIdx.x * blockDim.x + threadIdx.x;
  if (i < n) p[i] = v;
}

// ------------------------------------------------------------------- degree
__global__ void deg_accum_kernel(const int* __restrict__ dst,
                                 const float* __restrict__ w,
                                 float* __restrict__ deg, int E) {
  int e = blockIdx.x * blockDim.x + threadIdx.x;
  if (e < E) atomicAdd(&deg[dst[e]], w[e]);
}

__global__ void deg_final_kernel(const float* __restrict__ deg,
                                 float* __restrict__ dinv,
                                 float* __restrict__ invdeg, int n) {
  int i = blockIdx.x * blockDim.x + threadIdx.x;
  if (i < n) {
    float d = deg[i];
    dinv[i]   = rsqrtf(d);
    invdeg[i] = 1.0f / d;
  }
}

// --------------------------------------------------------------- WMMA GEMM
// C[M x Nc] = A[M x K] * B, f32 -> f16 fragments, f32 accumulation via
// v_wmma_f32_16x16x32_f16. One wave per 16x16 C tile. K is compile-time;
// out-of-range fragment elements use clamped-address loads + value select
// (v_cndmask) so no EXEC-masked load diamonds are generated.
// TRANSB=false: B is [K x Nc] row-major (B[k*ldb + n])
// TRANSB=true : B is [Nc x K] row-major (B[n*ldb + k]) == B^T, contiguous k.
template <bool TRANSB, int K>
__global__ void gemm_wmma_kernel(const float* __restrict__ A, int lda,
                                 const float* __restrict__ B, int ldb,
                                 float* __restrict__ C, int ldc,
                                 int M, int Ncols) {
  const int lane   = threadIdx.x & 31;
  const int wave   = blockIdx.x * (blockDim.x >> 5) + (threadIdx.x >> 5);
  const int tilesN = Ncols >> 4;
  const int nTiles = (M >> 4) * tilesN;
  if (wave >= nTiles) return;

  const int mBase = (wave / tilesN) << 4;
  const int nBase = (wave % tilesN) << 4;
  const int half  = lane >> 4;   // 0: lanes 0-15, 1: lanes 16-31
  const int l15   = lane & 15;

  const float* Arow = A + (size_t)(mBase + l15) * lda;
  const int    ncol = nBase + l15;
  const float* Bc   = TRANSB ? (B + (size_t)ncol * ldb) : (B + ncol);

  v8f acc = {};
#pragma unroll
  for (int k0 = 0; k0 < K; k0 += 32) {
    // ---- A fragment: 16x32 f16 (ISA 7.12.2). Per lane: two contiguous
    // 8-float runs at k0+8*half and k0+16+8*half -> b128-mergeable.
    v16h a;
#pragma unroll
    for (int c = 0; c < 2; ++c) {
      const int kb = k0 + c * 16 + half * 8;
#pragma unroll
      for (int t = 0; t < 8; ++t) {
        const int  k  = kb + t;
        const bool ok = (k < K);               // folds for K%32==0
        float x = Arow[ok ? k : 0];            // clamped, always in-bounds
        a[c * 8 + t] = (_Float16)(ok ? x : 0.0f);
      }
    }
    // ---- B fragment: 32x16 f16, same k order as A: element e holds
    // K = k0 + 16*(e/8) + 8*half + e%8.
    v16h b;
    const int kbB = k0 + half * 8;
#pragma unroll
    for (int e = 0; e < 16; ++e) {
      const int  k  = kbB + (e >> 3) * 16 + (e & 7);
      const bool ok = (k < K);
      const int  kc = ok ? k : 0;
      float x = TRANSB ? Bc[kc] : Bc[(size_t)kc * ldb];
      b[e] = (_Float16)(ok ? x : 0.0f);
    }
    acc = __builtin_amdgcn_wmma_f32_16x16x32_f16(
        /*neg_a=*/false, a, /*neg_b=*/false, b,
        /*c_mod=*/(short)0, acc, /*reuse_a=*/false, /*reuse_b=*/false);
  }
  // ---- store C tile: vgpr r holds M = mBase + r + 8*half, N = nBase + l15
#pragma unroll
  for (int r = 0; r < 8; ++r) {
    const int row = mBase + r + half * 8;
    C[(size_t)row * ldc + (nBase + l15)] = acc[r];
  }
}

// ------------------------------------------------- edge propagation (scatter)
// One wave per edge: 32 lanes x float2 cover the 64-wide feature row.
__global__ void propagate_kernel(const int* __restrict__ src,
                                 const int* __restrict__ dst,
                                 const float* __restrict__ w,
                                 const float* __restrict__ dinv,
                                 const float* __restrict__ h,   // [N x 64]
                                 float* __restrict__ agg,       // [N x 64]
                                 int E) {
  const int lane = threadIdx.x & 31;
  const int e    = blockIdx.x * (blockDim.x >> 5) + (threadIdx.x >> 5);
  if (e >= E) return;

  const int s = src[e];
  const int d = dst[e];
  const float norm = dinv[s] * w[e] * dinv[d];

  // prefetch a future edge's source row (global_prefetch_b8)
  if (e + 64 < E) {
    int s2 = src[e + 64];
    __builtin_prefetch(h + (size_t)s2 * NH + 2 * lane, 0, 0);
  }

  const float2* hs = (const float2*)(h + (size_t)s * NH);     // b64 load
  float* ad = agg + (size_t)d * NH + 2 * lane;
  float2 v = hs[lane];
  atomicAdd(ad,     v.x * norm);
  atomicAdd(ad + 1, v.y * norm);
}

// --------------------------------- self-loop + bias + ReLU + BN -> feats col
__global__ void gcn_finalize_kernel(const float* __restrict__ agg,
                                    const float* __restrict__ lin,
                                    const float* __restrict__ invdeg,
                                    const float* __restrict__ bias,
                                    const float* __restrict__ gamma,
                                    const float* __restrict__ beta,
                                    const float* __restrict__ mean,
                                    const float* __restrict__ var,
                                    float* __restrict__ feats,  // [N x 128]
                                    int colOff) {
  int i = blockIdx.x * blockDim.x + threadIdx.x;
  if (i >= NNODES * NH) return;
  int node = i >> 6;
  int j    = i & 63;
  float v = agg[i] + lin[i] * invdeg[node] + bias[j];
  v = fmaxf(v, 0.0f);
  v = (v - mean[j]) * rsqrtf(var[j] + BN_EPS) * gamma[j] + beta[j];
  feats[(size_t)node * 128 + colOff + j] = v;
}

// ------------------------------------------------------- LSTM step from zero
__device__ __forceinline__ float sigmoidf_(float x) {
  return 1.0f / (1.0f + __expf(-x));
}

__global__ void lstm_act_kernel(const float* __restrict__ gates,  // [N x 256]
                                const float* __restrict__ bih,
                                const float* __restrict__ bhh,
                                float* __restrict__ lh) {         // [N x 64]
  int i = blockIdx.x * blockDim.x + threadIdx.x;
  if (i >= NNODES * NH) return;
  int node = i >> 6;
  int j    = i & 63;
  const float* g = gates + (size_t)node * (4 * NH);
  float ig = sigmoidf_(g[j]          + bih[j]          + bhh[j]);
  float gg = tanhf    (g[2 * NH + j] + bih[2 * NH + j] + bhh[2 * NH + j]);
  float og = sigmoidf_(g[3 * NH + j] + bih[3 * NH + j] + bhh[3 * NH + j]);
  float c  = ig * gg;                      // f-gate * c0 == 0
  lh[i] = og * tanhf(c);
}

// ------------------------------------------------ final Linear(2H+F_IN -> 1)
__global__ void final_linear_kernel(const float* __restrict__ lh1,
                                    const float* __restrict__ lh2,
                                    const float* __restrict__ x,
                                    const float* __restrict__ Wlin,
                                    const float* __restrict__ blin,
                                    float* __restrict__ out) {
  int n = blockIdx.x * blockDim.x + threadIdx.x;
  if (n >= NNODES) return;
  float acc = blin[0];
  const float* a  = lh1 + (size_t)n * NH;
  const float* b  = lh2 + (size_t)n * NH;
  const float* xr = x   + (size_t)n * FINN;
#pragma unroll
  for (int j = 0; j < NH; ++j)   acc += fmaxf(a[j], 0.0f) * Wlin[j];
#pragma unroll
  for (int j = 0; j < NH; ++j)   acc += fmaxf(b[j], 0.0f) * Wlin[NH + j];
#pragma unroll
  for (int j = 0; j < FINN; ++j) acc += fmaxf(xr[j], 0.0f) * Wlin[2 * NH + j];
  out[n] = acc;
}

// ---------------------------------------------------------------------------
extern "C" void kernel_launch(void* const* d_in, const int* in_sizes, int n_in,
                              void* d_out, int out_size, void* d_ws, size_t ws_size,
                              hipStream_t stream) {
  const float* x    = (const float*)d_in[0];
  const int*   eidx = (const int*)  d_in[1];
  const float* ew   = (const float*)d_in[2];
  const float* W1   = (const float*)d_in[3];
  const float* b1   = (const float*)d_in[4];
  const float* W2   = (const float*)d_in[5];
  const float* b2   = (const float*)d_in[6];
  const float* bn1g = (const float*)d_in[7];
  const float* bn1b = (const float*)d_in[8];
  const float* bn1m = (const float*)d_in[9];
  const float* bn1v = (const float*)d_in[10];
  const float* bn2g = (const float*)d_in[11];
  const float* bn2b = (const float*)d_in[12];
  const float* bn2m = (const float*)d_in[13];
  const float* bn2v = (const float*)d_in[14];
  const float* Wih1 = (const float*)d_in[15];
  const float* bih1 = (const float*)d_in[16];
  const float* bhh1 = (const float*)d_in[17];
  const float* Wih2 = (const float*)d_in[18];
  const float* bih2 = (const float*)d_in[19];
  const float* bhh2 = (const float*)d_in[20];
  const float* Wlin = (const float*)d_in[21];
  const float* blin = (const float*)d_in[22];
  float* out = (float*)d_out;

  const int E = in_sizes[2];                 // edge count (edge_weight)
  const int* src = eidx;                     // edge_index[0]
  const int* dst = eidx + E;                 // edge_index[1]

  // -------- workspace carve-up (floats)
  float* ws     = (float*)d_ws;
  float* deg    = ws;                                // N
  float* dinv   = deg    + NNODES;                   // N
  float* invdeg = dinv   + NNODES;                   // N
  float* lin    = invdeg + NNODES;                   // N*64 (reused per layer)
  float* agg    = lin    + (size_t)NNODES * NH;      // N*64 (reused per layer)
  float* feats  = agg    + (size_t)NNODES * NH;      // N*128 (h1 | h2)
  float* gates  = feats  + (size_t)NNODES * 2 * NH;  // N*256 (reused)
  float* lh1    = gates  + (size_t)NNODES * 4 * NH;  // N*64
  float* lh2    = lh1    + (size_t)NNODES * NH;      // N*64

  const int T = 256;
  const int nBlkN  = (NNODES + T - 1) / T;
  const int nBlkE  = (E + T - 1) / T;
  const int nBlkNH = (NNODES * NH + T - 1) / T;
  const int nBlkEw = (E + 7) / 8;  // 8 waves/block, 1 edge per wave

  auto gemmBlocks = [](int M, int Nc) {
    int tiles = (M >> 4) * (Nc >> 4);
    return (tiles + 7) / 8;  // 8 waves (tiles) per 256-thread block
  };

  // -------- degree: deg = 1 + sum_w, then rsqrt / reciprocal
  hipLaunchKernelGGL(set_value_kernel, dim3(nBlkN), dim3(T), 0, stream,
                     deg, 1.0f, NNODES);
  hipLaunchKernelGGL(deg_accum_kernel, dim3(nBlkE), dim3(T), 0, stream,
                     dst, ew, deg, E);
  hipLaunchKernelGGL(deg_final_kernel, dim3(nBlkN), dim3(T), 0, stream,
                     deg, dinv, invdeg, NNODES);

  // -------- GCN layer 1: lin = x @ W1  (K=8 zero-padded to 32)
  hipLaunchKernelGGL((gemm_wmma_kernel<false, FINN>),
                     dim3(gemmBlocks(NNODES, NH)), dim3(T), 0, stream,
                     x, FINN, W1, NH, lin, NH, NNODES, NH);
  hipLaunchKernelGGL(set_value_kernel, dim3(nBlkNH), dim3(T), 0, stream,
                     agg, 0.0f, NNODES * NH);
  hipLaunchKernelGGL(propagate_kernel, dim3(nBlkEw), dim3(T), 0, stream,
                     src, dst, ew, dinv, lin, agg, E);
  hipLaunchKernelGGL(gcn_finalize_kernel, dim3(nBlkNH), dim3(T), 0, stream,
                     agg, lin, invdeg, b1, bn1g, bn1b, bn1m, bn1v, feats, 0);

  // -------- GCN layer 2: lin = h1 @ W2  (h1 = feats[:, :64], lda=128)
  hipLaunchKernelGGL((gemm_wmma_kernel<false, NH>),
                     dim3(gemmBlocks(NNODES, NH)), dim3(T), 0, stream,
                     feats, 2 * NH, W2, NH, lin, NH, NNODES, NH);
  hipLaunchKernelGGL(set_value_kernel, dim3(nBlkNH), dim3(T), 0, stream,
                     agg, 0.0f, NNODES * NH);
  hipLaunchKernelGGL(propagate_kernel, dim3(nBlkEw), dim3(T), 0, stream,
                     src, dst, ew, dinv, lin, agg, E);
  hipLaunchKernelGGL(gcn_finalize_kernel, dim3(nBlkNH), dim3(T), 0, stream,
                     agg, lin, invdeg, b2, bn2g, bn2b, bn2m, bn2v, feats, NH);

  // -------- LSTM 1: gates = feats @ Wih1^T  ([N,128] x [128,256])
  hipLaunchKernelGGL((gemm_wmma_kernel<true, 2 * NH>),
                     dim3(gemmBlocks(NNODES, 4 * NH)), dim3(T), 0, stream,
                     feats, 2 * NH, Wih1, 2 * NH, gates, 4 * NH,
                     NNODES, 4 * NH);
  hipLaunchKernelGGL(lstm_act_kernel, dim3(nBlkNH), dim3(T), 0, stream,
                     gates, bih1, bhh1, lh1);

  // -------- LSTM 2: gates = lh1 @ Wih2^T  ([N,64] x [64,256])
  hipLaunchKernelGGL((gemm_wmma_kernel<true, NH>),
                     dim3(gemmBlocks(NNODES, 4 * NH)), dim3(T), 0, stream,
                     lh1, NH, Wih2, NH, gates, 4 * NH, NNODES, 4 * NH);
  hipLaunchKernelGGL(lstm_act_kernel, dim3(nBlkNH), dim3(T), 0, stream,
                     gates, bih2, bhh2, lh2);

  // -------- out = relu([lh1|lh2|x]) @ Wlin + blin
  hipLaunchKernelGGL(final_linear_kernel, dim3(nBlkN), dim3(T), 0, stream,
                     lh1, lh2, x, Wlin, blin, out);
}
